// GatedParallelConvSSM_19628000543422
// MI455X (gfx1250) — compile-verified, matrix operand order checked by reading, not run on hardware
//
#include <hip/hip_runtime.h>
#include <math.h>

// ---------------------------------------------------------------------------
// Problem constants
// ---------------------------------------------------------------------------
constexpr int BATCH = 8;
constexpr int HH    = 128;
constexpr int WW    = 128;
constexpr int C     = 256;
constexpr int T_STEPS = 8;
constexpr int KS    = 7;
constexpr int M_ROWS = BATCH * HH * WW;          // 131072 GEMM rows
constexpr size_t N_ELEM = (size_t)M_ROWS * C;    // 33,554,432
constexpr size_t HWC    = (size_t)HH * WW * C;   // 4,194,304 (pow2)

#define PI_F 3.14159265358979323846f

// ---------------------------------------------------------------------------
// WMMA types (CDNA5 / gfx1250: wave32, V_WMMA_F32_16X16X32_BF16)
// ---------------------------------------------------------------------------
typedef __attribute__((ext_vector_type(16))) __bf16 v16bf;
typedef __attribute__((ext_vector_type(8)))  float  v8f;

union Frag {
  v16bf v;
  unsigned short h[16];
  uint4 q[2];
};

__device__ __forceinline__ unsigned short f32_to_bf16_rne(float f) {
  union { float f; unsigned u; } in; in.f = f;
  unsigned u = in.u;
  u += 0x7FFFu + ((u >> 16) & 1u);   // round-to-nearest-even
  return (unsigned short)(u >> 16);
}

// ---------------------------------------------------------------------------
// Weight swizzle: W[q] (256x256 f32, row-major [k,n]) -> bf16 B-fragments.
// Flat layout: (((q*16 + tn)*8 + k0)*32 + lane)*16 + e, so a wave's fragment
// load is two coalesced b128 per lane.  Total 512 KB -> L2-resident.
// Fragment mapping: lane = (g<<4)|nr,  e -> k = 32*k0 + 16*(e>>3) + 8*g + (e&7)
// ---------------------------------------------------------------------------
__global__ void swizzle_W(const float* __restrict__ W0, const float* __restrict__ W1,
                          const float* __restrict__ W2, const float* __restrict__ W3,
                          unsigned short* __restrict__ Bsw) {
  int o = blockIdx.x * 256 + threadIdx.x;        // 4*16*8*32*16 = 262144 total
  int e    = o & 15;
  int lane = (o >> 4) & 31;
  int k0   = (o >> 9) & 7;
  int tn   = (o >> 12) & 15;
  int q    = o >> 16;
  const float* W = (q == 0) ? W0 : (q == 1) ? W1 : (q == 2) ? W2 : W3;
  int k = k0 * 32 + ((e >> 3) << 4) + ((lane >> 4) << 3) + (e & 7);
  int n = tn * 16 + (lane & 15);
  Bsw[o] = f32_to_bf16_rne(W[(size_t)k * C + n]);
}

// ---------------------------------------------------------------------------
// Fused 4-way gate GEMM.  One wave owns one 16-row m-tile; its A panel
// (16x256) is converted to bf16 fragments ONCE (held in 64 VGPRs) and reused
// across all 4 weights x 16 n-tiles.  x is read exactly once from HBM.
// ---------------------------------------------------------------------------
__device__ __forceinline__ void gemm_one_weight(const Frag a[8],
                                                const unsigned short* __restrict__ bq,
                                                const float* __restrict__ bias,
                                                float* __restrict__ y,
                                                int mt, int lane, int g, int mr) {
  #pragma unroll 1
  for (int tn = 0; tn < 16; ++tn) {
    // lane's B pointer for this n-tile: 32 contiguous bytes per (k0,lane)
    const unsigned short* bp = bq + ((size_t)tn * 8 * 32 + lane) * 16;
    __builtin_prefetch(bq + ((size_t)(tn + 1) * 8 * 32 + lane) * 16, 0, 1);
    v8f acc = {0.f, 0.f, 0.f, 0.f, 0.f, 0.f, 0.f, 0.f};
    #pragma unroll
    for (int k0 = 0; k0 < 8; ++k0) {
      Frag b;
      const uint4* bv = (const uint4*)(bp + (size_t)k0 * 32 * 16);
      b.q[0] = bv[0];
      b.q[1] = bv[1];
      acc = __builtin_amdgcn_wmma_f32_16x16x32_bf16(
          false, a[k0].v, false, b.v, (short)0, acc, false, false);
    }
    // C/D layout: lane -> n = lane&15 ; VGPR r -> m = r + 8*(lane>>4)
    int n = tn * 16 + mr;
    float bvadd = bias[n];
    #pragma unroll
    for (int r = 0; r < 8; ++r)
      y[(size_t)(mt * 16 + r + 8 * g) * C + n] = acc[r] + bvadd;
  }
}

__global__ void gates_gemm_fused(const float* __restrict__ x,
                                 const unsigned short* __restrict__ Bsw,
                                 const float* __restrict__ b0, const float* __restrict__ b1,
                                 const float* __restrict__ b2, const float* __restrict__ b3,
                                 float* __restrict__ y0, float* __restrict__ y1,
                                 float* __restrict__ y2, float* __restrict__ y3) {
  const int lane = threadIdx.x & 31;
  const int wave = threadIdx.x >> 5;             // 8 waves / block
  const int mt   = blockIdx.x * 8 + wave;        // m-tile id (8192 total)
  const int g    = lane >> 4;
  const int mr   = lane & 15;

  // Stage A fragments: 16x256 panel -> 8 bf16 fragments in registers.
  // element e of frag k0 -> k = 32*k0 + 16*(e>>3) + 8*g + (e&7):
  // two contiguous 8-float runs per fragment -> four b128 loads + cvt.
  const float* __restrict__ xrow = x + (size_t)(mt * 16 + mr) * C;
  Frag a[8];
  #pragma unroll
  for (int k0 = 0; k0 < 8; ++k0) {
    #pragma unroll
    for (int hf = 0; hf < 2; ++hf) {             // e = 8*hf + j
      const float4* p = (const float4*)(xrow + k0 * 32 + 16 * hf + 8 * g);
      float4 f0 = p[0];
      float4 f1 = p[1];
      a[k0].h[8 * hf + 0] = f32_to_bf16_rne(f0.x);
      a[k0].h[8 * hf + 1] = f32_to_bf16_rne(f0.y);
      a[k0].h[8 * hf + 2] = f32_to_bf16_rne(f0.z);
      a[k0].h[8 * hf + 3] = f32_to_bf16_rne(f0.w);
      a[k0].h[8 * hf + 4] = f32_to_bf16_rne(f1.x);
      a[k0].h[8 * hf + 5] = f32_to_bf16_rne(f1.y);
      a[k0].h[8 * hf + 6] = f32_to_bf16_rne(f1.z);
      a[k0].h[8 * hf + 7] = f32_to_bf16_rne(f1.w);
    }
  }

  const size_t wstride = (size_t)16 * 8 * 32 * 16;   // shorts per weight matrix
  gemm_one_weight(a, Bsw + 0 * wstride, b0, y0, mt, lane, g, mr);
  gemm_one_weight(a, Bsw + 1 * wstride, b1, y1, mt, lane, g, mr);
  gemm_one_weight(a, Bsw + 2 * wstride, b2, y2, mt, lane, g, mr);
  gemm_one_weight(a, Bsw + 3 * wstride, b3, y3, mt, lane, g, mr);
}

// ---------------------------------------------------------------------------
// Direct 49-tap DFT of the wrap-padded 7x7 kernels -> A_f, B_f  (h,w,c)
// ---------------------------------------------------------------------------
__global__ void kernel_freq(const float* __restrict__ Ak,
                            const float* __restrict__ Bk,
                            float2* __restrict__ Af,
                            float2* __restrict__ Bf) {
  const int uv = blockIdx.x;
  const int u = uv >> 7;
  const int v = uv & 127;
  const int c = threadIdx.x;

  __shared__ float2 tw[KS * KS];
  if (c < KS * KS) {
    int di = c / KS - KS / 2;
    int dj = c % KS - KS / 2;
    float ang = -2.0f * PI_F * (float)(u * di + v * dj) / (float)HH;
    tw[c] = make_float2(__cosf(ang), __sinf(ang));
  }
  __syncthreads();

  float2 sa = make_float2(0.f, 0.f);
  float2 sb = make_float2(0.f, 0.f);
  #pragma unroll
  for (int t = 0; t < KS * KS; ++t) {
    float2 w = tw[t];
    float ka = Ak[(size_t)c * (KS * KS) + t];
    float kb = Bk[(size_t)c * (KS * KS) + t];
    sa.x += ka * w.x; sa.y += ka * w.y;
    sb.x += kb * w.x; sb.y += kb * w.y;
  }
  size_t o = (size_t)uv * C + c;
  Af[o] = sa;
  Bf[o] = sb;
}

// ---------------------------------------------------------------------------
// Radix-2 DIT 128-point FFT over 32 channels in LDS (conflict-free b64).
// ---------------------------------------------------------------------------
constexpr int CB = 32;

__device__ __forceinline__ void fft128_stages(float2* lds, const float2* tw, int tid) {
  #pragma unroll
  for (int s = 0; s < 7; ++s) {
    const int half = 1 << s;
    for (int t = tid; t < 64 * CB; t += 256) {
      int j   = t >> 5;
      int c   = t & (CB - 1);
      int pos = j & (half - 1);
      int i0  = ((j >> s) << (s + 1)) + pos;
      int i1  = i0 + half;
      float2 Wt = tw[pos << (6 - s)];
      float2 uu = lds[i0 * CB + c];
      float2 vv = lds[i1 * CB + c];
      float2 vw = make_float2(vv.x * Wt.x - vv.y * Wt.y,
                              vv.x * Wt.y + vv.y * Wt.x);
      lds[i0 * CB + c] = make_float2(uu.x + vw.x, uu.y + vw.y);
      lds[i1 * CB + c] = make_float2(uu.x - vw.x, uu.y - vw.y);
    }
    __syncthreads();
  }
}

__device__ __forceinline__ void build_tw(float2* tw, int tid, float dirsign) {
  if (tid < 64) {
    float a = dirsign * 2.0f * PI_F * (float)tid / 128.0f;
    tw[tid] = make_float2(__cosf(a), __sinf(a));
  }
}

__global__ void fft_w_r2c(const float* __restrict__ x, float2* __restrict__ xf) {
  __shared__ float2 lds[128 * CB];
  __shared__ float2 tw[64];
  const int tid  = threadIdx.x;
  const int cblk = blockIdx.x & 7;
  const int bh   = blockIdx.x >> 3;
  build_tw(tw, tid, -1.0f);

  const size_t base = (size_t)bh * WW * C + (size_t)cblk * CB;
  for (int idx = tid; idx < 128 * CB; idx += 256) {
    int p = idx >> 5, c = idx & (CB - 1);
    int pr = __brev((unsigned)p) >> 25;
    lds[pr * CB + c] = make_float2(x[base + (size_t)p * C + c], 0.0f);
  }
  __syncthreads();
  fft128_stages(lds, tw, tid);
  for (int idx = tid; idx < 128 * CB; idx += 256) {
    int p = idx >> 5, c = idx & (CB - 1);
    xf[base + (size_t)p * C + c] = lds[p * CB + c];
  }
}

__global__ void fft_pass(float2* __restrict__ xf, int dimH, float dirsign) {
  __shared__ float2 lds[128 * CB];
  __shared__ float2 tw[64];
  const int tid  = threadIdx.x;
  const int cblk = blockIdx.x & 7;
  const int o    = blockIdx.x >> 3;
  build_tw(tw, tid, dirsign);

  size_t base, stride;
  if (dimH) {
    int b = o >> 7, w = o & 127;
    base   = ((size_t)b * HH * WW + w) * C + (size_t)cblk * CB;
    stride = (size_t)WW * C;
  } else {
    base   = (size_t)o * WW * C + (size_t)cblk * CB;
    stride = C;
  }

  for (int idx = tid; idx < 128 * CB; idx += 256) {
    int p = idx >> 5, c = idx & (CB - 1);
    int pr = __brev((unsigned)p) >> 25;
    lds[pr * CB + c] = xf[base + (size_t)p * stride + c];
  }
  __syncthreads();
  fft128_stages(lds, tw, tid);
  for (int idx = tid; idx < 128 * CB; idx += 256) {
    int p = idx >> 5, c = idx & (CB - 1);
    xf[base + (size_t)p * stride + c] = lds[p * CB + c];
  }
}

__global__ void ifft_h_final(const float2* __restrict__ xf, float* __restrict__ out) {
  __shared__ float2 lds[128 * CB];
  __shared__ float2 tw[64];
  const int tid  = threadIdx.x;
  const int cblk = blockIdx.x & 7;
  const int o    = blockIdx.x >> 3;
  build_tw(tw, tid, +1.0f);

  const int b = o >> 7, w = o & 127;
  const size_t base   = ((size_t)b * HH * WW + w) * C + (size_t)cblk * CB;
  const size_t stride = (size_t)WW * C;

  for (int idx = tid; idx < 128 * CB; idx += 256) {
    int p = idx >> 5, c = idx & (CB - 1);
    int pr = __brev((unsigned)p) >> 25;
    lds[pr * CB + c] = xf[base + (size_t)p * stride + c];
  }
  __syncthreads();
  fft128_stages(lds, tw, tid);
  const float scale = 1.0f / (float)(HH * WW);
  for (int idx = tid; idx < 128 * CB; idx += 256) {
    int p = idx >> 5, c = idx & (CB - 1);
    out[base + (size_t)p * stride + c] = lds[p * CB + c].x * scale;
  }
}

// ---------------------------------------------------------------------------
// Fused gating + geometric recurrence (in-place on x_f)
// ---------------------------------------------------------------------------
__device__ __forceinline__ float2 cmul(float2 a, float2 b) {
  return make_float2(a.x * b.x - a.y * b.y, a.x * b.y + a.y * b.x);
}

__global__ void combine(const float* __restrict__ pma, const float* __restrict__ ppa,
                        const float* __restrict__ pmb, const float* __restrict__ ppb,
                        const float2* __restrict__ Af, const float2* __restrict__ Bf,
                        float2* __restrict__ xf) {
  size_t i  = (size_t)blockIdx.x * 256 + threadIdx.x;
  size_t sp = i & (HWC - 1);

  float ma = 1.0f / (1.0f + __expf(-pma[i]));
  float mb = 1.0f / (1.0f + __expf(-pmb[i]));
  float sa, ca, sb, cb;
  __sincosf(ppa[i], &sa, &ca);
  __sincosf(ppb[i], &sb, &cb);
  float2 ga = make_float2(ma * ca, ma * sa);
  float2 gb = make_float2(mb * cb, mb * sb);

  float2 a  = cmul(ga, Af[sp]);
  float2 bb = cmul(cmul(gb, Bf[sp]), xf[i]);

  float2 h = bb;
  #pragma unroll
  for (int t = 1; t < T_STEPS; ++t)
    h = make_float2(a.x * h.x - a.y * h.y + bb.x,
                    a.x * h.y + a.y * h.x + bb.y);
  xf[i] = h;
}

// ---------------------------------------------------------------------------
// Launcher
// ---------------------------------------------------------------------------
extern "C" void kernel_launch(void* const* d_in, const int* in_sizes, int n_in,
                              void* d_out, int out_size, void* d_ws, size_t ws_size,
                              hipStream_t stream) {
  const float* x    = (const float*)d_in[0];
  const float* Ak   = (const float*)d_in[1];
  const float* Bk   = (const float*)d_in[2];
  const float* W_ma = (const float*)d_in[3];
  const float* b_ma = (const float*)d_in[4];
  const float* W_pa = (const float*)d_in[5];
  const float* b_pa = (const float*)d_in[6];
  const float* W_mb = (const float*)d_in[7];
  const float* b_mb = (const float*)d_in[8];
  const float* W_pb = (const float*)d_in[9];
  const float* b_pb = (const float*)d_in[10];
  float* out = (float*)d_out;

  // Workspace carve-up (~832 MiB + 512 KiB)
  char* ws = (char*)d_ws;
  float* p_ma = (float*)ws;  ws += N_ELEM * sizeof(float);
  float* p_pa = (float*)ws;  ws += N_ELEM * sizeof(float);
  float* p_mb = (float*)ws;  ws += N_ELEM * sizeof(float);
  float* p_pb = (float*)ws;  ws += N_ELEM * sizeof(float);
  float2* xf  = (float2*)ws; ws += N_ELEM * sizeof(float2);
  float2* Af  = (float2*)ws; ws += HWC * sizeof(float2);
  float2* Bf  = (float2*)ws; ws += HWC * sizeof(float2);
  unsigned short* Bsw = (unsigned short*)ws;   // 4*16*8*32*16 shorts = 512 KiB

  // 1) Swizzle weights into WMMA B-fragment layout (bf16, L2-resident)
  swizzle_W<<<1024, 256, 0, stream>>>(W_ma, W_pa, W_mb, W_pb, Bsw);

  // 2) Fused 4-way gate GEMM: x read once, 512 WMMA per wave
  gates_gemm_fused<<<M_ROWS / 16 / 8, 256, 0, stream>>>(
      x, Bsw, b_ma, b_pa, b_mb, b_pb, p_ma, p_pa, p_mb, p_pb);

  // 3) Conv kernels -> frequency domain
  kernel_freq<<<HH * WW, 256, 0, stream>>>(Ak, Bk, Af, Bf);

  // 4) Forward 2D FFT of x
  const int fft_blocks = BATCH * HH * (C / CB);   // 8192
  fft_w_r2c<<<fft_blocks, 256, 0, stream>>>(x, xf);
  fft_pass<<<fft_blocks, 256, 0, stream>>>(xf, 1, -1.0f);

  // 5) Gating + T-step geometric recurrence (in place)
  combine<<<(unsigned)(N_ELEM / 256), 256, 0, stream>>>(p_ma, p_pa, p_mb, p_pb,
                                                        Af, Bf, xf);

  // 6) Inverse 2D FFT, real part to output
  fft_pass<<<fft_blocks, 256, 0, stream>>>(xf, 0, +1.0f);
  ifft_h_final<<<fft_blocks, 256, 0, stream>>>(xf, out);
}
